// PLSTMModel_57638461112616
// MI455X (gfx1250) — matrix operand order repaired and checked by reading, not compiled, not run on hardware
//
#include <hip/hip_runtime.h>
#include <math.h>

#define B_   256
#define P_   5
#define T_   256
#define F_   128
#define H_   512
#define C_   60
#define D_   (P_ * F_)     /* 640  */
#define G4H_ (4 * H_)      /* 2048 */
#define RON_   0.05f
#define ALPHA_ 0.001f

typedef _Float16 v8h  __attribute__((ext_vector_type(8)));
typedef _Float16 v16h __attribute__((ext_vector_type(16)));
typedef float    v8f  __attribute__((ext_vector_type(8)));

// ---------------------------------------------------------------------------
// WMMA fragment loaders (wave32, V_WMMA_F32_16X16X32_F16 layouts)
// A (16x32, row-major src, stride lda):
//   lane = m + 16*kmid ; element e = (k%8) + 8*(k/16), kmid = (k%16)/8
//   -> per lane: two contiguous 8-half (16B) loads at k0+8*kmid and +16.
// B (32x16 from W^T rows, stride ldk):
//   lane = n + 16*kmid ; element e = (k%8) + 8*(k/16), kmid = (k%16)/8
//   -> per lane: two contiguous 8-half (16B) loads at k0+8*kmid and +16.
// ---------------------------------------------------------------------------
__device__ __forceinline__ v16h load_a_frag(const _Float16* __restrict__ A,
                                            int lda, int row0, int k0, int lane) {
  const int m    = lane & 15;
  const int kmid = lane >> 4;
  const _Float16* p = A + (size_t)(row0 + m) * lda + k0 + 8 * kmid;
  v8h lo = *(const v8h*)(p);
  v8h hi = *(const v8h*)(p + 16);
  return __builtin_shufflevector(lo, hi, 0,1,2,3,4,5,6,7,8,9,10,11,12,13,14,15);
}

__device__ __forceinline__ v16h load_b_frag(const _Float16* __restrict__ WT,
                                            int ldk, int col0, int k0, int lane) {
  const int n    = lane & 15;
  const int kgrp = lane >> 4;
  const _Float16* p = WT + (size_t)(col0 + n) * ldk + k0 + 8 * kgrp;
  v8h lo = *(const v8h*)(p);
  v8h hi = *(const v8h*)(p + 16);
  return __builtin_shufflevector(lo, hi, 0,1,2,3,4,5,6,7,8,9,10,11,12,13,14,15);
}

__device__ __forceinline__ v8f wmma_f16(v16h a, v16h b, v8f c) {
  return __builtin_amdgcn_wmma_f32_16x16x32_f16(false, a, false, b,
                                                (short)0, c, false, false);
}

// ---------------------------------------------------------------------------
// Pack inputs (B,P,T,F) f32 -> X (T, B, D) f16 (time-major, D = P*F)
// ---------------------------------------------------------------------------
__global__ void k_pack_x(const float* __restrict__ in, _Float16* __restrict__ X) {
  size_t tid = (size_t)blockIdx.x * blockDim.x + threadIdx.x;
  if (tid >= (size_t)T_ * B_ * D_) return;
  int f = (int)(tid % F_);
  size_t r = tid / F_;
  int p = (int)(r % P_); r /= P_;
  int b = (int)(r % B_);
  int t = (int)(r / B_);
  float v = in[(((size_t)b * P_ + p) * T_ + t) * F_ + f];
  X[tid] = (_Float16)v;  // tid == ((t*B + b)*D + p*F + f)
}

// W (K x N) f32 -> WT (N x K) f16
__global__ void k_pack_wT(const float* __restrict__ W, _Float16* __restrict__ WT,
                          int K, int N) {
  size_t tid = (size_t)blockIdx.x * blockDim.x + threadIdx.x;
  if (tid >= (size_t)K * N) return;
  int k = (int)(tid % K);
  int n = (int)(tid / K);
  WT[(size_t)n * K + k] = (_Float16)W[(size_t)k * N + n];
}

__global__ void k_zero_state(float* c, float* h0, float* h1,
                             _Float16* hh0, _Float16* hh1) {
  int tid = blockIdx.x * blockDim.x + threadIdx.x;
  if (tid < B_ * H_) {
    c[tid]  = 0.f; h0[tid] = 0.f; h1[tid] = 0.f;
    hh0[tid] = (_Float16)0.f; hh1[tid] = (_Float16)0.f;
  }
}

// ---------------------------------------------------------------------------
// Phase 1: G = X(65536x640) @ WxT^T + bias  -> f16 (65536 x 2048)
// Workgroup: 256 thr = 8 waves, tile 128x128; wave = 32x64 (2x4 WMMA tiles).
// ---------------------------------------------------------------------------
__global__ __launch_bounds__(256, 1)
void k_gemm_xw(const _Float16* __restrict__ X,   // M x D
               const _Float16* __restrict__ WxT, // 4H x D
               const float*    __restrict__ bias,
               _Float16*       __restrict__ G) { // M x 4H
  const int lane = threadIdx.x & 31;
  const int wave = threadIdx.x >> 5;
  const int row0 = blockIdx.y * 128 + (wave & 3) * 32;
  const int col0 = blockIdx.x * 128 + (wave >> 2) * 64;

  v8f acc[2][4];
  #pragma unroll
  for (int i = 0; i < 2; i++)
    #pragma unroll
    for (int j = 0; j < 4; j++)
      #pragma unroll
      for (int e = 0; e < 8; e++) acc[i][j][e] = 0.f;

  for (int k0 = 0; k0 < D_; k0 += 32) {
    v16h a[2], b[4];
    #pragma unroll
    for (int i = 0; i < 2; i++) a[i] = load_a_frag(X, D_, row0 + 16 * i, k0, lane);
    #pragma unroll
    for (int j = 0; j < 4; j++) b[j] = load_b_frag(WxT, D_, col0 + 16 * j, k0, lane);
    #pragma unroll
    for (int i = 0; i < 2; i++)
      #pragma unroll
      for (int j = 0; j < 4; j++)
        acc[i][j] = wmma_f16(a[i], b[j], acc[i][j]);
  }

  const int rlane = lane >> 4;
  const int nlane = lane & 15;
  #pragma unroll
  for (int i = 0; i < 2; i++) {
    #pragma unroll
    for (int j = 0; j < 4; j++) {
      const int col = col0 + 16 * j + nlane;
      const float bb = bias[col];
      #pragma unroll
      for (int r = 0; r < 8; r++) {
        const int row = row0 + 16 * i + 8 * rlane + r;
        G[(size_t)row * G4H_ + col] = (_Float16)(acc[i][j][r] + bb);
      }
    }
  }
}

// ---------------------------------------------------------------------------
// Phase 2: one fused step. gates = G[t] + h @ Wh, then LSTM + phased gate.
// Grid (8,4): block tile = 64 batch rows x 64 hidden cols x 4 gates.
// Wave = 1 m-tile x 2 hidden n-tiles x 4 gates = 8 accum sets, K=512.
// ---------------------------------------------------------------------------
__global__ __launch_bounds__(256, 1)
void k_step(const _Float16* __restrict__ G,      // (T*B) x 4H
            const _Float16* __restrict__ WhT,    // 4H x H
            const _Float16* __restrict__ h_in16, // B x H
            const float*    __restrict__ h_in32, // B x H
            const float*    __restrict__ tau,
            const float*    __restrict__ shift,
            float*          __restrict__ c,       // B x H (in/out)
            float*          __restrict__ h_out32,
            _Float16*       __restrict__ h_out16,
            int t) {
  const int lane  = threadIdx.x & 31;
  const int wave  = threadIdx.x >> 5;
  const int row0  = blockIdx.y * 64 + (wave & 3) * 16;  // batch rows
  const int hcol0 = blockIdx.x * 64 + (wave >> 2) * 32; // hidden cols (2 tiles)
  const int rlane = lane >> 4;
  const int nlane = lane & 15;

  const _Float16* Gt = G + (size_t)t * B_ * G4H_;

  v8f acc[2][4];  // [hidden n-tile][gate i,f,g,o]
  #pragma unroll
  for (int nt = 0; nt < 2; nt++)
    #pragma unroll
    for (int gt = 0; gt < 4; gt++) {
      const int col = gt * H_ + hcol0 + 16 * nt + nlane;
      #pragma unroll
      for (int r = 0; r < 8; r++) {
        const int row = row0 + 8 * rlane + r;
        acc[nt][gt][r] = (float)Gt[(size_t)row * G4H_ + col];
      }
    }

  for (int k0 = 0; k0 < H_; k0 += 32) {
    v16h a = load_a_frag(h_in16, H_, row0, k0, lane);
    #pragma unroll
    for (int nt = 0; nt < 2; nt++)
      #pragma unroll
      for (int gt = 0; gt < 4; gt++) {
        v16h b = load_b_frag(WhT, H_, gt * H_ + hcol0 + 16 * nt, k0, lane);
        acc[nt][gt] = wmma_f16(a, b, acc[nt][gt]);
      }
  }

  #pragma unroll
  for (int nt = 0; nt < 2; nt++) {
    const int h = hcol0 + 16 * nt + nlane;
    // phased-LSTM time gate k(t, h)
    const float tv = tau[h];
    float phi = fmodf((float)t - shift[h], tv);
    if (phi < 0.f) phi += tv;
    phi /= tv;
    float kg;
    if (phi < 0.5f * RON_)      kg = 2.f * phi / RON_;
    else if (phi < RON_)        kg = 2.f - 2.f * phi / RON_;
    else                        kg = ALPHA_ * phi;

    #pragma unroll
    for (int r = 0; r < 8; r++) {
      const int row = row0 + 8 * rlane + r;
      const size_t idx = (size_t)row * H_ + h;
      const float ig = 1.f / (1.f + expf(-acc[nt][0][r]));
      const float fg = 1.f / (1.f + expf(-acc[nt][1][r]));
      const float gg = tanhf(acc[nt][2][r]);
      const float og = 1.f / (1.f + expf(-acc[nt][3][r]));
      const float cv = c[idx];
      const float ct = fg * cv + ig * gg;
      const float ht = og * tanhf(ct);
      const float cn = kg * ct + (1.f - kg) * cv;
      const float hn = kg * ht + (1.f - kg) * h_in32[idx];
      c[idx]       = cn;
      h_out32[idx] = hn;
      h_out16[idx] = (_Float16)hn;
    }
  }
}

// ---------------------------------------------------------------------------
// Final: logits = hT @ fc_w + fc_b ; log_softmax over C=60. One block per row.
// ---------------------------------------------------------------------------
__global__ void k_fc_logsoftmax(const float* __restrict__ h,
                                const float* __restrict__ w,
                                const float* __restrict__ bias,
                                float* __restrict__ out) {
  __shared__ float ls[C_];
  __shared__ float red[2];
  const int b = blockIdx.x;
  const int j = threadIdx.x;
  if (j < C_) {
    float acc = bias[j];
    for (int k = 0; k < H_; k++)
      acc += h[(size_t)b * H_ + k] * w[(size_t)k * C_ + j];
    ls[j] = acc;
  }
  __syncthreads();
  if (j == 0) {
    float m = ls[0];
    for (int i = 1; i < C_; i++) m = fmaxf(m, ls[i]);
    float s = 0.f;
    for (int i = 0; i < C_; i++) s += expf(ls[i] - m);
    red[0] = m;
    red[1] = logf(s);
  }
  __syncthreads();
  if (j < C_) out[(size_t)b * C_ + j] = ls[j] - red[0] - red[1];
}

// ---------------------------------------------------------------------------
extern "C" void kernel_launch(void* const* d_in, const int* in_sizes, int n_in,
                              void* d_out, int out_size, void* d_ws, size_t ws_size,
                              hipStream_t stream) {
  (void)in_sizes; (void)n_in; (void)out_size; (void)ws_size;
  const float* inputs = (const float*)d_in[0];
  const float* Wx     = (const float*)d_in[1];
  const float* Wh     = (const float*)d_in[2];
  const float* bias   = (const float*)d_in[3];
  const float* tau    = (const float*)d_in[4];
  const float* shift  = (const float*)d_in[5];
  const float* fcw    = (const float*)d_in[6];
  const float* fcb    = (const float*)d_in[7];
  float* out = (float*)d_out;

  char* ws = (char*)d_ws;
  size_t off = 0;
  auto alloc = [&](size_t bytes) -> void* {
    void* p = ws + off;
    off += (bytes + 255) & ~(size_t)255;
    return p;
  };
  _Float16* X    = (_Float16*)alloc((size_t)T_ * B_ * D_ * 2);    //  84 MB
  _Float16* WxT  = (_Float16*)alloc((size_t)G4H_ * D_ * 2);       // 2.6 MB
  _Float16* WhT  = (_Float16*)alloc((size_t)G4H_ * H_ * 2);       // 2.1 MB
  _Float16* G    = (_Float16*)alloc((size_t)T_ * B_ * G4H_ * 2);  // 268 MB
  float*    c    = (float*)   alloc((size_t)B_ * H_ * 4);
  float*    h32a = (float*)   alloc((size_t)B_ * H_ * 4);
  float*    h32b = (float*)   alloc((size_t)B_ * H_ * 4);
  _Float16* h16a = (_Float16*)alloc((size_t)B_ * H_ * 2);
  _Float16* h16b = (_Float16*)alloc((size_t)B_ * H_ * 2);
  float*    h32[2]  = { h32a, h32b };
  _Float16* h16[2]  = { h16a, h16b };

  // pack / transpose / zero
  {
    size_t nx = (size_t)T_ * B_ * D_;
    k_pack_x<<<(unsigned)((nx + 255) / 256), 256, 0, stream>>>(inputs, X);
    size_t nwx = (size_t)D_ * G4H_;
    k_pack_wT<<<(unsigned)((nwx + 255) / 256), 256, 0, stream>>>(Wx, WxT, D_, G4H_);
    size_t nwh = (size_t)H_ * G4H_;
    k_pack_wT<<<(unsigned)((nwh + 255) / 256), 256, 0, stream>>>(Wh, WhT, H_, G4H_);
    k_zero_state<<<(B_ * H_ + 255) / 256, 256, 0, stream>>>(c, h32[0], h32[1], h16[0], h16[1]);
  }

  // Phase 1: hoisted x-GEMM (fully parallel, saturates WMMA)
  {
    dim3 grid(G4H_ / 128, (T_ * B_) / 128);  // 16 x 512 = 8192 workgroups
    k_gemm_xw<<<grid, 256, 0, stream>>>(X, WxT, bias, G);
  }

  // Phase 2: 256 fused recurrent steps (ping-pong h buffers)
  {
    dim3 grid(H_ / 64, B_ / 64);  // 8 x 4 = 32 workgroups per step
    for (int t = 0; t < T_; t++) {
      const int pi = t & 1, po = pi ^ 1;
      k_step<<<grid, 256, 0, stream>>>(G, WhT, h16[pi], h32[pi], tau, shift,
                                       c, h32[po], h16[po], t);
    }
  }

  // Final FC + log_softmax; last write went to buffer (T_ % 2) == 0
  k_fc_logsoftmax<<<B_, 64, 0, stream>>>(h32[T_ % 2], fcw, fcb, out);
}
